// LightweightTemporalTransformer_10582799417585
// MI455X (gfx1250) — matrix operand order repaired and checked by reading, not compiled, not run on hardware
//
#include <hip/hip_runtime.h>
#include <hip/hip_bf16.h>

typedef _Float16 half_t;
typedef __attribute__((ext_vector_type(16))) _Float16 v16h;
typedef __attribute__((ext_vector_type(8)))  _Float16 v8h_t;
typedef __attribute__((ext_vector_type(8)))  float    v8f;

#define Dm   256
#define SEQ  2048
#define NROW 8192   // B * S
#define NH   4
#define HDM  64

__device__ __forceinline__ v8f wmma16(v16h a, v16h b, v8f c) {
  return __builtin_amdgcn_wmma_f32_16x16x32_f16(false, a, false, b, (short)0, c, false, false);
}

// A operand: 16x32 f16 tile from row-major [M][K] (leading dim ld, elements).
// Per ISA: lane<16 holds K[k0+0..7] and K[k0+16..23]; lane>=16 holds K[k0+8..15], K[k0+24..31].
__device__ __forceinline__ v16h load_fragA(const half_t* __restrict__ base, int m0, int ld,
                                           int k0, int lane) {
  int r  = lane & 15;
  int hl = lane >> 4;
  const half_t* p = base + (size_t)(m0 + r) * ld + k0 + hl * 8;
  v8h_t lo = *(const v8h_t*)(p);
  v8h_t hi = *(const v8h_t*)(p + 16);
  v16h a;
#pragma unroll
  for (int i = 0; i < 8; ++i) { a[i] = lo[i]; a[i + 8] = hi[i]; }
  return a;
}

// B operand: 32x16 (KxN) tile, source stored row-major [N][K] (transposed weights).
// Per ISA: lanes 0-15 hold K[k0+0..7] and K[k0+16..23] for N=lane; lanes 16-31 hold K[k0+8..15], K[k0+24..31].
__device__ __forceinline__ v16h load_fragB(const half_t* __restrict__ base, int n0, int ld,
                                           int k0, int lane) {
  int r  = lane & 15;
  int hl = lane >> 4;
  const half_t* p = base + (size_t)(n0 + r) * ld + k0 + hl * 8;
  v8h_t lo = *(const v8h_t*)(p);
  v8h_t hi = *(const v8h_t*)(p + 16);
  v16h b;
#pragma unroll
  for (int i = 0; i < 8; ++i) { b[i] = lo[i]; b[i + 8] = hi[i]; }
  return b;
}

// ---------- conversion kernels ----------
__global__ void k_cvt_x(const float* __restrict__ x, half_t* __restrict__ xh) {
  int i = blockIdx.x * blockDim.x + threadIdx.x;
  xh[i] = (half_t)x[i];
}

__global__ void k_cvt_wT(const float* __restrict__ w, half_t* __restrict__ wt) {
  int nn = blockIdx.x, kk = threadIdx.x;   // wt[n][k] = w[k][n], 256x256
  wt[nn * Dm + kk] = (half_t)w[kk * Dm + nn];
}

// ---------- GEMM (+bias) fused with LayerNorm; writes residual f32 and LN f16 ----------
__global__ __launch_bounds__(512) void k_gemm_ln(const half_t* __restrict__ A,
    const half_t* __restrict__ Wt, const float* __restrict__ bias,
    const float* __restrict__ g, const float* __restrict__ be,
    float* __restrict__ resOut, half_t* __restrict__ hOut) {
  __shared__ float L[16][Dm + 4];
  int tid = threadIdx.x;
  int wave = tid >> 5, lane = tid & 31;
  int row0 = blockIdx.x * 16;
  int col0 = wave * 16;
  v8f c = {};
#pragma unroll
  for (int k = 0; k < Dm; k += 32) {
    v16h a = load_fragA(A, row0, Dm, k, lane);
    v16h b = load_fragB(Wt, col0, Dm, k, lane);
    c = wmma16(a, b, c);
  }
  int n = lane & 15, hl = lane >> 4;
  float bcol = bias[col0 + n];
#pragma unroll
  for (int j = 0; j < 8; ++j) L[8 * hl + j][col0 + n] = c[j] + bcol;
  __syncthreads();
  int r = wave;                       // wave w normalizes row w
  float v[8];
  float s = 0.f, s2 = 0.f;
#pragma unroll
  for (int i = 0; i < 8; ++i) { v[i] = L[r][lane * 8 + i]; s += v[i]; s2 += v[i] * v[i]; }
#pragma unroll
  for (int m = 16; m >= 1; m >>= 1) { s += __shfl_xor(s, m, 32); s2 += __shfl_xor(s2, m, 32); }
  float mu = s * (1.f / Dm);
  float rstd = rsqrtf(s2 * (1.f / Dm) - mu * mu + 1e-5f);
  size_t ro = (size_t)(row0 + r) * Dm;
#pragma unroll
  for (int i = 0; i < 8; ++i) {
    int cc = lane * 8 + i;
    float xv = v[i];
    resOut[ro + cc] = xv;
    hOut[ro + cc] = (half_t)((xv - mu) * rstd * g[cc] + be[cc]);
  }
}

// ---------- QKV GEMM (+bias) fused with RoPE; blockIdx.y selects q/k/v ----------
__global__ __launch_bounds__(512) void k_gemm_qkv(const half_t* __restrict__ A,
    const half_t* __restrict__ WqT, const half_t* __restrict__ WkT, const half_t* __restrict__ WvT,
    const float* __restrict__ bq, const float* __restrict__ bk, const float* __restrict__ bv,
    half_t* __restrict__ qOut, half_t* __restrict__ kOut, half_t* __restrict__ vtOut) {
  __shared__ float L[16][Dm + 4];
  int sel = blockIdx.y;
  const half_t* Wt   = (sel == 0) ? WqT : (sel == 1 ? WkT : WvT);
  const float*  bias = (sel == 0) ? bq  : (sel == 1 ? bk  : bv);
  int tid = threadIdx.x;
  int wave = tid >> 5, lane = tid & 31;
  int row0 = blockIdx.x * 16;
  int col0 = wave * 16;
  v8f c = {};
#pragma unroll
  for (int k = 0; k < Dm; k += 32) {
    v16h a = load_fragA(A, row0, Dm, k, lane);
    v16h b = load_fragB(Wt, col0, Dm, k, lane);
    c = wmma16(a, b, c);
  }
  int n = lane & 15, hl = lane >> 4;
  float bcol = bias[col0 + n];
#pragma unroll
  for (int j = 0; j < 8; ++j) L[8 * hl + j][col0 + n] = c[j] + bcol;
  __syncthreads();
  int bb = row0 >> 11;              // batch
  int s0 = row0 & (SEQ - 1);        // sequence start of strip
  if (sel == 2) {
    // V: store transposed [bh][64][S] so PV B-operand loads are contiguous in keys
#pragma unroll
    for (int e = 0; e < 8; ++e) {
      int idx = tid + 512 * e;
      int r = idx >> 8, cc = idx & 255;
      int head = cc >> 6, d = cc & 63;
      vtOut[(((size_t)(bb * NH + head)) * HDM + d) * SEQ + (s0 + r)] = (half_t)L[r][cc];
    }
  } else {
    half_t* outp = (sel == 0) ? qOut : kOut;
#pragma unroll
    for (int e = 0; e < 8; ++e) {
      int idx = tid + 512 * e;
      int r = idx >> 8, cc = idx & 255;
      int head = cc >> 6, d = cc & 63;
      int j = d & 31;
      // inv_freq[j] = 10000^(-2j/64) = exp(-ln(1e4)/32 * j)
      float fr = __expf(-0.28782313662425575f * (float)j);
      float ang = (float)(s0 + r) * fr;
      float sn, cs;
      __sincosf(ang, &sn, &cs);
      float t0 = L[r][(head << 6) + 2 * j];
      float t1 = L[r][(head << 6) + 2 * j + 1];
      float val = (d < 32) ? (t0 * cs - t1 * sn) : (t1 * cs + t0 * sn);
      outp[(((size_t)(bb * NH + head)) * SEQ + (s0 + r)) * HDM + d] = (half_t)val;
    }
  }
}

// ---------- flash attention: one wave per (head, 16-query strip) ----------
__global__ __launch_bounds__(32) void k_attn(const half_t* __restrict__ qh,
    const half_t* __restrict__ kh, const half_t* __restrict__ vt,
    const int* __restrict__ mask, half_t* __restrict__ aoh) {
  __shared__ __align__(16) half_t Pst[16][32];   // C-layout -> A-layout staging
  int lane = threadIdx.x;
  int bh = blockIdx.y;
  int bb = bh >> 2, hh = bh & 3;
  int q0 = blockIdx.x * 16;
  const half_t* Q = qh + (size_t)bh * SEQ * HDM;
  const half_t* K = kh + (size_t)bh * SEQ * HDM;
  const half_t* V = vt + (size_t)bh * HDM * SEQ;
  v16h qa0 = load_fragA(Q, q0, HDM, 0, lane);    // Q stays resident all loop
  v16h qa1 = load_fragA(Q, q0, HDM, 32, lane);
  int n = lane & 15, hl = lane >> 4;
  float rowscale[8], mrow[8], lrow[8];
#pragma unroll
  for (int j = 0; j < 8; ++j) {
    // masked query rows: force scores to 0 -> softmax becomes uniform -> out = mean(V)
    rowscale[j] = mask[bb * SEQ + q0 + 8 * hl + j] ? 0.125f : 0.0f;  // 0.125 = HD^-0.5
    mrow[j] = -1e30f;
    lrow[j] = 0.f;
  }
  v8f acc[4] = {};
  for (int kt = 0; kt < SEQ; kt += 32) {
    v8f sa = {}, sb = {};
    sa = wmma16(qa0, load_fragB(K, kt,      HDM, 0,  lane), sa);
    sa = wmma16(qa1, load_fragB(K, kt,      HDM, 32, lane), sa);
    sb = wmma16(qa0, load_fragB(K, kt + 16, HDM, 0,  lane), sb);
    sb = wmma16(qa1, load_fragB(K, kt + 16, HDM, 32, lane), sb);
    float corr[8];
#pragma unroll
    for (int j = 0; j < 8; ++j) {
      float va = sa[j] * rowscale[j];
      float vb = sb[j] * rowscale[j];
      float mx = fmaxf(va, vb);
#pragma unroll
      for (int msk = 8; msk >= 1; msk >>= 1) mx = fmaxf(mx, __shfl_xor(mx, msk, 32));
      float nm = fmaxf(mrow[j], mx);
      float co = __expf(mrow[j] - nm);
      float pa = __expf(va - nm);
      float pb = __expf(vb - nm);
      float ps = pa + pb;
#pragma unroll
      for (int msk = 8; msk >= 1; msk >>= 1) ps += __shfl_xor(ps, msk, 32);
      lrow[j] = lrow[j] * co + ps;
      mrow[j] = nm;
      corr[j] = co;
      Pst[8 * hl + j][n]      = (half_t)pa;      // row = M, col = key-in-tile
      Pst[8 * hl + j][16 + n] = (half_t)pb;
    }
    asm volatile("s_wait_dscnt 0" ::: "memory"); // same-wave cross-lane LDS handoff
    v16h pf = load_fragA(&Pst[0][0], 0, 32, 0, lane);
#pragma unroll
    for (int t = 0; t < 4; ++t) {
#pragma unroll
      for (int j = 0; j < 8; ++j) acc[t][j] *= corr[j];
      acc[t] = wmma16(pf, load_fragB(V, 16 * t, SEQ, kt, lane), acc[t]);
    }
  }
#pragma unroll
  for (int t = 0; t < 4; ++t) {
#pragma unroll
    for (int j = 0; j < 8; ++j) {
      int qrow = q0 + 8 * hl + j;
      int col = hh * HDM + 16 * t + n;
      aoh[((size_t)bb * SEQ + qrow) * Dm + col] = (half_t)(acc[t][j] / lrow[j]);
    }
  }
}

// ---------- out-proj GEMM + bias + residual, fused LN, y = x2 + LN(x2) ----------
__global__ __launch_bounds__(512) void k_gemm_out_ln(const half_t* __restrict__ A,
    const half_t* __restrict__ Wt, const float* __restrict__ bias,
    const float* __restrict__ resIn, const float* __restrict__ g,
    const float* __restrict__ be, float* __restrict__ out) {
  __shared__ float L[16][Dm + 4];
  int tid = threadIdx.x;
  int wave = tid >> 5, lane = tid & 31;
  int row0 = blockIdx.x * 16;
  int col0 = wave * 16;
  v8f c = {};
#pragma unroll
  for (int k = 0; k < Dm; k += 32) {
    v16h a = load_fragA(A, row0, Dm, k, lane);
    v16h b = load_fragB(Wt, col0, Dm, k, lane);
    c = wmma16(a, b, c);
  }
  int n = lane & 15, hl = lane >> 4;
  float bcol = bias[col0 + n];
#pragma unroll
  for (int j = 0; j < 8; ++j) {
    int rr = 8 * hl + j;
    L[rr][col0 + n] = c[j] + bcol + resIn[(size_t)(row0 + rr) * Dm + col0 + n];
  }
  __syncthreads();
  int r = wave;
  float v[8];
  float s = 0.f, s2 = 0.f;
#pragma unroll
  for (int i = 0; i < 8; ++i) { v[i] = L[r][lane * 8 + i]; s += v[i]; s2 += v[i] * v[i]; }
#pragma unroll
  for (int m = 16; m >= 1; m >>= 1) { s += __shfl_xor(s, m, 32); s2 += __shfl_xor(s2, m, 32); }
  float mu = s * (1.f / Dm);
  float rstd = rsqrtf(s2 * (1.f / Dm) - mu * mu + 1e-5f);
  size_t ro = (size_t)(row0 + r) * Dm;
#pragma unroll
  for (int i = 0; i < 8; ++i) {
    int cc = lane * 8 + i;
    float xv = v[i];
    out[ro + cc] = xv + (xv - mu) * rstd * g[cc] + be[cc];
  }
}

extern "C" void kernel_launch(void* const* d_in, const int* in_sizes, int n_in,
                              void* d_out, int out_size, void* d_ws, size_t ws_size,
                              hipStream_t stream) {
  (void)in_sizes; (void)n_in; (void)out_size; (void)ws_size;
  const float* x    = (const float*)d_in[0];
  const int*   mask = (const int*)  d_in[1];
  const float* w_in = (const float*)d_in[2];
  const float* b_in = (const float*)d_in[3];
  const float* g1   = (const float*)d_in[4];
  const float* be1  = (const float*)d_in[5];
  const float* wq   = (const float*)d_in[6];
  const float* bq   = (const float*)d_in[7];
  const float* wk   = (const float*)d_in[8];
  const float* bk   = (const float*)d_in[9];
  const float* wv   = (const float*)d_in[10];
  const float* bv   = (const float*)d_in[11];
  const float* wo   = (const float*)d_in[12];
  const float* bo   = (const float*)d_in[13];
  const float* g2   = (const float*)d_in[14];
  const float* be2  = (const float*)d_in[15];
  // d_in[16..21] (MoE params) are dead: reference multiplies that branch by 0.0
  float* out = (float*)d_out;
  char* ws = (char*)d_ws;

  size_t off = 0;
  auto take = [&](size_t bytes) { size_t r = off; off += (bytes + 255) & ~(size_t)255; return r; };
  half_t* xh   = (half_t*)(ws + take((size_t)NROW * Dm * 2));
  half_t* hh   = (half_t*)(ws + take((size_t)NROW * Dm * 2));
  float*  resb = (float*) (ws + take((size_t)NROW * Dm * 4));
  half_t* wiT  = (half_t*)(ws + take((size_t)Dm * Dm * 2));
  half_t* wqT  = (half_t*)(ws + take((size_t)Dm * Dm * 2));
  half_t* wkT  = (half_t*)(ws + take((size_t)Dm * Dm * 2));
  half_t* wvT  = (half_t*)(ws + take((size_t)Dm * Dm * 2));
  half_t* woT  = (half_t*)(ws + take((size_t)Dm * Dm * 2));
  half_t* qhB  = (half_t*)(ws + take((size_t)16 * SEQ * HDM * 2));
  half_t* khB  = (half_t*)(ws + take((size_t)16 * SEQ * HDM * 2));
  half_t* vtB  = (half_t*)(ws + take((size_t)16 * SEQ * HDM * 2));
  half_t* aoh  = (half_t*)(ws + take((size_t)NROW * Dm * 2));

  k_cvt_x<<<(NROW * Dm) / 256, 256, 0, stream>>>(x, xh);
  k_cvt_wT<<<Dm, Dm, 0, stream>>>(w_in, wiT);
  k_cvt_wT<<<Dm, Dm, 0, stream>>>(wq, wqT);
  k_cvt_wT<<<Dm, Dm, 0, stream>>>(wk, wkT);
  k_cvt_wT<<<Dm, Dm, 0, stream>>>(wv, wvT);
  k_cvt_wT<<<Dm, Dm, 0, stream>>>(wo, woT);

  k_gemm_ln<<<NROW / 16, 512, 0, stream>>>(xh, wiT, b_in, g1, be1, resb, hh);
  k_gemm_qkv<<<dim3(NROW / 16, 3), 512, 0, stream>>>(hh, wqT, wkT, wvT, bq, bk, bv, qhB, khB, vtB);
  k_attn<<<dim3(SEQ / 16, 16), 32, 0, stream>>>(qhB, khB, vtB, mask, aoh);
  k_gemm_out_ln<<<NROW / 16, 512, 0, stream>>>(aoh, woT, bo, resb, g2, be2, out);
}
